// Attention_84104049590786
// MI455X (gfx1250) — matrix operand order repaired and checked by reading, not compiled, not run on hardware
//
#include <hip/hip_runtime.h>
#include <hip/hip_bf16.h>
#include <math.h>

// Problem constants (B, N, DM, H, DH) = (4, 2048, 1024, 16, 64)
#define BATCH 4
#define SEQ   2048
#define DMODEL 1024
#define NHEAD 16
#define DHEAD 64
#define QKVC  3072              // 3*H*DH
#define SCALE 0.125f            // DH^-0.5
#define MASKV 1000000.0f

typedef __bf16 bf16_t;
typedef __attribute__((ext_vector_type(16))) __bf16 v16bf;
typedef __attribute__((ext_vector_type(8)))  __bf16 v8bf;
typedef __attribute__((ext_vector_type(8)))  float  v8f;
typedef __attribute__((ext_vector_type(4)))  float  v4f;

// ---------------------------------------------------------------------------
// WMMA helpers: D = A(16x32 bf16) * B(32x16 bf16) + C(16x16 f32)
// ---------------------------------------------------------------------------
__device__ __forceinline__ v8f wmma_bf16(v16bf a, v16bf b, v8f c) {
    return __builtin_amdgcn_wmma_f32_16x16x32_bf16(
        false, a, false, b, (short)0, c, false, false);
}

__device__ __forceinline__ v16bf pack16(v8bf lo, v8bf hi) {
    v16bf r;
#pragma unroll
    for (int i = 0; i < 8; ++i) { r[i] = lo[i]; r[i + 8] = hi[i]; }
    return r;
}

// A fragment (16x32, MxK) from a row-major bf16 tile, contiguous-k layout (k = 16*half + s):
// lane(0-15)=row M, elems = K (lanes 16-31 take K=16..31) -> 16 contiguous.
__device__ __forceinline__ v16bf load_a_bf16(const bf16_t* p, int lda, int lane) {
    const int m = lane & 15, half = lane >> 4;
    const bf16_t* r = p + (size_t)m * lda + half * 16;
    v8bf lo = *(const v8bf*)(r);
    v8bf hi = *(const v8bf*)(r + 8);
    return pack16(lo, hi);
}

// B fragment (32x16, KxN) from an N-major (row=n, len-K) bf16 tile:
// lane(0-15)=col N, elems = K (lanes 16-31 take K=16..31) -> 16 contiguous.
__device__ __forceinline__ v16bf load_b_nk(const bf16_t* p, int ldb, int lane) {
    const int n = lane & 15, half = lane >> 4;
    const bf16_t* r = p + (size_t)n * ldb + half * 16;
    v8bf lo = *(const v8bf*)(r);
    v8bf hi = *(const v8bf*)(r + 8);
    return pack16(lo, hi);
}

// Reductions across a 16-lane group (rows of a C fragment live across lanes).
__device__ __forceinline__ v8f red16_max(v8f x) {
#pragma unroll
    for (int m = 1; m < 16; m <<= 1)
#pragma unroll
        for (int v = 0; v < 8; ++v) x[v] = fmaxf(x[v], __shfl_xor(x[v], m, 32));
    return x;
}
__device__ __forceinline__ v8f red16_sum(v8f x) {
#pragma unroll
    for (int m = 1; m < 16; m <<= 1)
#pragma unroll
        for (int v = 0; v < 8; ++v) x[v] += __shfl_xor(x[v], m, 32);
    return x;
}

// ---------------------------------------------------------------------------
// Tensor Data Mover: 2-D tile (tile_k x tile_n bf16) global -> LDS.
// D# layout per CDNA5 ISA (type=2, data_size=1 -> 2B).
// ---------------------------------------------------------------------------
#if __has_builtin(__builtin_amdgcn_tensor_load_to_lds)
#define HAVE_TDM 1
typedef __attribute__((ext_vector_type(4))) unsigned int u32x4;
typedef __attribute__((ext_vector_type(8))) int i32x8;
typedef __attribute__((ext_vector_type(4))) int i32x4;

__device__ __forceinline__ void tdm_load_b_tile(unsigned lds_off,
                                                const bf16_t* gtile,
                                                int tile_k, int tile_n,
                                                int tensor_k, int tensor_n,
                                                int stride_k) {
    const unsigned long long ga = (unsigned long long)(uintptr_t)gtile;
    u32x4 g0;
    g0[0] = 1u;                                      // count=1 user descriptor
    g0[1] = lds_off;                                 // LDS byte address
    g0[2] = (unsigned)(ga & 0xffffffffu);            // global_addr[31:0]
    g0[3] = (unsigned)(ga >> 32) | 0x80000000u;      // addr[56:32] | type=2
    i32x8 g1;
    g1[0] = (int)(1u << 16);                         // data_size=1 (2 bytes)
    g1[1] = (int)((unsigned)tensor_k << 16);         // tensor_dim0[15:0]
    g1[2] = (int)(((unsigned)tensor_k >> 16) | ((unsigned)tensor_n << 16));
    g1[3] = (int)(((unsigned)tensor_n >> 16) | ((unsigned)tile_k << 16));
    g1[4] = (int)(unsigned)tile_n;                   // tile_dim1 (tile_dim2=0)
    g1[5] = (int)(unsigned)stride_k;                 // tensor_dim0_stride lo
    g1[6] = 0;
    g1[7] = 0;
    i32x4 z4 = {};
#if defined(__clang_major__) && (__clang_major__ >= 23)
    i32x8 z8 = {};
    __builtin_amdgcn_tensor_load_to_lds(g0, g1, z4, z4, z8, 0);
#else
    __builtin_amdgcn_tensor_load_to_lds(g0, g1, z4, z4, 0);
#endif
}
#else
#define HAVE_TDM 0
#endif

// ---------------------------------------------------------------------------
// Kernel 0a: elementwise fp32 -> bf16 (for x; bandwidth-bound one-shot pass
// that removes v_cvt chains from the GEMM critical path).
// ---------------------------------------------------------------------------
__global__ void __launch_bounds__(256)
convert_f32_bf16(const float* __restrict__ src, bf16_t* __restrict__ dst,
                 int n8 /* n/8 */) {
    const int i = blockIdx.x * 256 + threadIdx.x;
    if (i >= n8) return;
    v4f a = *(const v4f*)(src + (size_t)i * 8);
    v4f b = *(const v4f*)(src + (size_t)i * 8 + 4);
    v8bf o;
#pragma unroll
    for (int v = 0; v < 4; ++v) { o[v] = (bf16_t)a[v]; o[v + 4] = (bf16_t)b[v]; }
    *(v8bf*)(dst + (size_t)i * 8) = o;
}

// ---------------------------------------------------------------------------
// Kernel 0b: fp32 (rows x cols) -> bf16 transposed (cols x rows), LDS-tiled.
// ---------------------------------------------------------------------------
__global__ void __launch_bounds__(256)
transpose_to_bf16(const float* __restrict__ src, bf16_t* __restrict__ dst,
                  int rows, int cols) {
    __shared__ float tile[32][33];
    const int tx = threadIdx.x & 31, ty = threadIdx.x >> 5;
    const int rt = rows >> 5;
    const int r0 = (blockIdx.x % rt) << 5;
    const int c0 = (blockIdx.x / rt) << 5;
#pragma unroll
    for (int i = 0; i < 4; ++i)
        tile[ty + i * 8][tx] = src[(size_t)(r0 + ty + i * 8) * cols + c0 + tx];
    __syncthreads();
#pragma unroll
    for (int i = 0; i < 4; ++i)
        dst[(size_t)(c0 + ty + i * 8) * rows + r0 + tx] =
            (bf16_t)tile[tx][ty + i * 8];
}

// ---------------------------------------------------------------------------
// GEMM B-tile staging: the whole block's 128-col x 32-k bf16 weight tile goes
// through LDS (TDM double-buffered when available, cooperative copy else).
// ---------------------------------------------------------------------------
#define BT_ELEMS (128 * 32)

__device__ __forceinline__ void stage_b_fallback(bf16_t* dst, const bf16_t* src) {
#pragma unroll
    for (int i = 0; i < 2; ++i) {
        const int chunk = threadIdx.x + i * 256;     // 512 chunks of 8 elems
        const int row = chunk >> 2, koff = (chunk & 3) * 8;
        *(v8bf*)(dst + row * 32 + koff) =
            *(const v8bf*)(src + (size_t)row * DMODEL + koff);
    }
}

// ---------------------------------------------------------------------------
// Kernel 1: QKV projection.  C[8192 x 3072] = x_bf16 @ w_qkv (B pre-transposed
// N-major bf16).  Wave tile 32x64 (2 A-frags x 4 B-frags = 8 WMMA / k-step).
// Block = 8 waves (4 M x 2 N) = 128 rows x 128 cols.  Epilogue scatters into
// Q [b,h,n,d], K [b,h,n,d], V^T [b,h,d,n].
// ---------------------------------------------------------------------------
__global__ void __launch_bounds__(256)
qkv_gemm(const bf16_t* __restrict__ xb, const bf16_t* __restrict__ wT,
         bf16_t* __restrict__ Qb, bf16_t* __restrict__ Kb,
         bf16_t* __restrict__ Vt) {
    __shared__ __align__(16) bf16_t ldsB[2][BT_ELEMS];

    const int lane = threadIdx.x & 31, wave = threadIdx.x >> 5;
    const int half = lane >> 4;
    const int mw = wave >> 1, nw = wave & 1;
    const int bx = blockIdx.x & 63;            // 64 M tiles of 128 rows
    const int by = blockIdx.x >> 6;            // 24 N tiles of 128 cols
    const int row0 = bx * 128 + mw * 32;
    const int colB = by * 128;                 // block column base

    v8f acc[2][4];
#pragma unroll
    for (int r = 0; r < 2; ++r)
#pragma unroll
        for (int f = 0; f < 4; ++f) acc[r][f] = (v8f){};

    const bf16_t* arow  = xb + (size_t)row0 * DMODEL;
    const bf16_t* btile = wT + (size_t)colB * DMODEL;

#if HAVE_TDM
    if (threadIdx.x < 32)                       // wave 0 drives the TDM
        tdm_load_b_tile((unsigned)(uintptr_t)&ldsB[0][0], btile,
                        32, 128, DMODEL, QKVC, DMODEL);
#endif
    for (int k0 = 0; k0 < DMODEL; k0 += 32) {
        const int buf = (k0 >> 5) & 1;
#if HAVE_TDM
        if (threadIdx.x < 32) {
            if (k0 + 32 < DMODEL) {
                tdm_load_b_tile((unsigned)(uintptr_t)&ldsB[buf ^ 1][0],
                                btile + k0 + 32, 32, 128, DMODEL, QKVC, DMODEL);
                __builtin_amdgcn_s_wait_tensorcnt(1);
            } else {
                __builtin_amdgcn_s_wait_tensorcnt(0);
            }
        }
#else
        stage_b_fallback(&ldsB[buf][0], btile + k0);
#endif
        __syncthreads();

        if (k0 + 64 < DMODEL)
            __builtin_prefetch(arow + k0 + 64, 0, 1);    // global_prefetch_b8
        // Batch all fragment loads, then the WMMA chain (latency overlap).
        v16bf a0 = load_a_bf16(arow + k0, DMODEL, lane);
        v16bf a1 = load_a_bf16(arow + 16 * DMODEL + k0, DMODEL, lane);
        v16bf bfr[4];
#pragma unroll
        for (int f = 0; f < 4; ++f)
            bfr[f] = load_b_nk(&ldsB[buf][(nw * 64 + f * 16) * 32], 32, lane);
#pragma unroll
        for (int f = 0; f < 4; ++f) acc[0][f] = wmma_bf16(a0, bfr[f], acc[0][f]);
#pragma unroll
        for (int f = 0; f < 4; ++f) acc[1][f] = wmma_bf16(a1, bfr[f], acc[1][f]);
        __syncthreads();
    }

    // Scatter.  Column c -> head h = c/192, which = (c%192)/64, d = c%64.
#pragma unroll
    for (int f = 0; f < 4; ++f) {
        const int cbase = colB + nw * 64 + f * 16;
        const int h     = cbase / 192;
        const int rem   = cbase - h * 192;
        const int which = rem / 64;
        const int d     = (rem - which * 64) + (lane & 15);
#pragma unroll
        for (int r = 0; r < 2; ++r)
#pragma unroll
            for (int v = 0; v < 8; ++v) {
                const int row = row0 + r * 16 + half * 8 + v;
                const int b   = row >> 11;
                const int n   = row & (SEQ - 1);
                const bf16_t val = (bf16_t)acc[r][f][v];
                const size_t bh = (size_t)(b * NHEAD + h);
                if (which == 0)      Qb[(bh * SEQ + n) * DHEAD + d] = val;
                else if (which == 1) Kb[(bh * SEQ + n) * DHEAD + d] = val;
                else                 Vt[(bh * DHEAD + d) * SEQ + n] = val;
            }
    }
}

// ---------------------------------------------------------------------------
// Kernel 2: flash attention.  One wave per 16-query tile, 4 waves/block.
// 32 keys/iter: 4 WMMA for S = Q K^T, online softmax in C-fragment registers,
// P transposed to A-layout via a 1KB LDS bounce (s_wait_dscnt fence),
// 4 WMMA for O += P V.  V fragment loads hoisted above the softmax VALU.
// ---------------------------------------------------------------------------
__global__ void __launch_bounds__(128)
flash_attn(const bf16_t* __restrict__ Qb, const bf16_t* __restrict__ Kb,
           const bf16_t* __restrict__ Vt, const int* __restrict__ mask,
           bf16_t* __restrict__ aout) {
    __shared__ __align__(16) bf16_t lds_p[4][16 * 32];

    const int lane = threadIdx.x & 31, wave = threadIdx.x >> 5;
    const int half = lane >> 4;
    const int nqt  = SEQ / 64;
    const int bh   = blockIdx.x / nqt;
    const int qt   = blockIdx.x % nqt;
    const int b    = bh >> 4, h = bh & 15;
    const int q0   = qt * 64 + wave * 16;

    const bf16_t* Qp = Qb + (size_t)bh * SEQ * DHEAD;
    const bf16_t* Kp = Kb + (size_t)bh * SEQ * DHEAD;
    const bf16_t* Vp = Vt + (size_t)bh * DHEAD * SEQ;
    const int*    mr = mask + b * SEQ;

    v16bf aq0 = load_a_bf16(Qp + (size_t)q0 * DHEAD,      DHEAD, lane);
    v16bf aq1 = load_a_bf16(Qp + (size_t)q0 * DHEAD + 32, DHEAD, lane);

    int mq[8];
#pragma unroll
    for (int v = 0; v < 8; ++v) mq[v] = mr[q0 + half * 8 + v];

    v8f m_run, l_run, o[4];
#pragma unroll
    for (int v = 0; v < 8; ++v) { m_run[v] = -3.0e38f; l_run[v] = 0.0f; }
#pragma unroll
    for (int f = 0; f < 4; ++f) o[f] = (v8f){};

    bf16_t* lp = lds_p[wave];

    for (int j0 = 0; j0 < SEQ; j0 += 32) {
        // ---- batch K fragments + mask, then score WMMAs ----
        v16bf bk[2][2];
        int mk[2];
#pragma unroll
        for (int kt = 0; kt < 2; ++kt) {
            const bf16_t* kp = Kp + (size_t)(j0 + kt * 16) * DHEAD;
            bk[kt][0] = load_b_nk(kp,      DHEAD, lane);
            bk[kt][1] = load_b_nk(kp + 32, DHEAD, lane);
            mk[kt] = mr[j0 + kt * 16 + (lane & 15)];
        }
        v8f s[2];
#pragma unroll
        for (int kt = 0; kt < 2; ++kt) {
            v8f c = (v8f){};
            c = wmma_bf16(aq0, bk[kt][0], c);
            c = wmma_bf16(aq1, bk[kt][1], c);
#pragma unroll
            for (int v = 0; v < 8; ++v)
                s[kt][v] = (mk[kt] && mq[v]) ? c[v] * SCALE : -MASKV;
        }

        // ---- hoist V fragment loads: in flight during softmax VALU ----
        v16bf bv[4];
#pragma unroll
        for (int f = 0; f < 4; ++f)
            bv[f] = load_b_nk(Vp + (size_t)(f * 16) * SEQ + j0, SEQ, lane);

        // ---- online softmax (row stats VGPR-element aligned with O) ----
        v8f t;
#pragma unroll
        for (int v = 0; v < 8; ++v) t[v] = fmaxf(s[0][v], s[1][v]);
        t = red16_max(t);
        v8f newm, corr, p0, p1, ps;
#pragma unroll
        for (int v = 0; v < 8; ++v) {
            newm[v] = fmaxf(m_run[v], t[v]);
            corr[v] = __expf(m_run[v] - newm[v]);
            p0[v]   = __expf(s[0][v] - newm[v]);
            p1[v]   = __expf(s[1][v] - newm[v]);
            ps[v]   = p0[v] + p1[v];
        }
        ps = red16_sum(ps);
#pragma unroll
        for (int v = 0; v < 8; ++v) l_run[v] = l_run[v] * corr[v] + ps[v];
#pragma unroll
        for (int f = 0; f < 4; ++f)
#pragma unroll
            for (int v = 0; v < 8; ++v) o[f][v] *= corr[v];
        m_run = newm;

        // ---- transpose P (C layout -> A layout) through LDS ----
#pragma unroll
        for (int v = 0; v < 8; ++v) {
            const int m = half * 8 + v;
            lp[m * 32 + (lane & 15)]      = (bf16_t)p0[v];
            lp[m * 32 + 16 + (lane & 15)] = (bf16_t)p1[v];
        }
        asm volatile("s_wait_dscnt 0" ::: "memory");
        __builtin_amdgcn_wave_barrier();
        v16bf ap = load_a_bf16(lp, 32, lane);

#pragma unroll
        for (int f = 0; f < 4; ++f) o[f] = wmma_bf16(ap, bv[f], o[f]);
        __builtin_amdgcn_wave_barrier();
    }

#pragma unroll
    for (int f = 0; f < 4; ++f) {
        const int col = h * DHEAD + f * 16 + (lane & 15);
#pragma unroll
        for (int v = 0; v < 8; ++v) {
            const int q = q0 + half * 8 + v;
            aout[((size_t)(b * SEQ + q)) * DMODEL + col] =
                (bf16_t)(o[f][v] / l_run[v]);
        }
    }
}

// ---------------------------------------------------------------------------
// Kernel 3: output projection.  out[8192 x 1024] = aout @ w_out + b_out (f32).
// Same TDM-staged structure as qkv_gemm, fp32 epilogue with bias.
// ---------------------------------------------------------------------------
__global__ void __launch_bounds__(256)
out_gemm(const bf16_t* __restrict__ A, const bf16_t* __restrict__ wT,
         const float* __restrict__ bias, float* __restrict__ out) {
    __shared__ __align__(16) bf16_t ldsB[2][BT_ELEMS];

    const int lane = threadIdx.x & 31, wave = threadIdx.x >> 5;
    const int half = lane >> 4;
    const int mw = wave >> 1, nw = wave & 1;
    const int bx = blockIdx.x & 63;
    const int by = blockIdx.x >> 6;            // 8 N tiles of 128 cols
    const int row0 = bx * 128 + mw * 32;
    const int colB = by * 128;

    v8f acc[2][4];
#pragma unroll
    for (int r = 0; r < 2; ++r)
#pragma unroll
        for (int f = 0; f < 4; ++f) acc[r][f] = (v8f){};

    const bf16_t* arow  = A + (size_t)row0 * DMODEL;
    const bf16_t* btile = wT + (size_t)colB * DMODEL;

#if HAVE_TDM
    if (threadIdx.x < 32)
        tdm_load_b_tile((unsigned)(uintptr_t)&ldsB[0][0], btile,
                        32, 128, DMODEL, DMODEL, DMODEL);
#endif
    for (int k0 = 0; k0 < DMODEL; k0 += 32) {
        const int buf = (k0 >> 5) & 1;
#if HAVE_TDM
        if (threadIdx.x < 32) {
            if (k0 + 32 < DMODEL) {
                tdm_load_b_tile((unsigned)(uintptr_t)&ldsB[buf ^ 1][0],
                                btile + k0 + 32, 32, 128, DMODEL, DMODEL, DMODEL);
                __builtin_amdgcn_s_wait_tensorcnt(1);
            } else {
                __builtin_amdgcn_s_wait_tensorcnt(0);
            }
        }
#else
        stage_b_fallback(&ldsB[buf][0], btile + k0);
#endif
        __syncthreads();

        if (k0 + 64 < DMODEL)
            __builtin_prefetch(arow + k0 + 64, 0, 1);
        v16bf a0 = load_a_bf16(arow + k0, DMODEL, lane);
        v16bf a1 = load_a_bf16(arow + 16 * DMODEL + k0, DMODEL, lane);
        v16bf bfr[4];
#pragma unroll
        for (int f = 0; f < 4; ++f)
            bfr[f] = load_b_nk(&ldsB[buf][(nw * 64 + f * 16) * 32], 32, lane);
#pragma unroll
        for (int f = 0; f < 4; ++f) acc[0][f] = wmma_bf16(a0, bfr[f], acc[0][f]);
#pragma unroll
        for (int f = 0; f < 4; ++f) acc[1][f] = wmma_bf16(a1, bfr[f], acc[1][f]);
        __syncthreads();
    }

#pragma unroll
    for (int f = 0; f < 4; ++f) {
        const int c  = colB + nw * 64 + f * 16 + (lane & 15);
        const float bv = bias[c];
#pragma unroll
        for (int r = 0; r < 2; ++r)
#pragma unroll
            for (int v = 0; v < 8; ++v)
                out[(size_t)(row0 + r * 16 + half * 8 + v) * DMODEL + c] =
                    acc[r][f][v] + bv;
    }
}

// ---------------------------------------------------------------------------
// Launch.  Workspace (bf16): wqkvT | woutT | x_bf16 | Q | K | V^T | aout
// (~92 MB)
// ---------------------------------------------------------------------------
extern "C" void kernel_launch(void* const* d_in, const int* in_sizes, int n_in,
                              void* d_out, int out_size, void* d_ws,
                              size_t ws_size, hipStream_t stream) {
    const float* x     = (const float*)d_in[0];
    const float* w_qkv = (const float*)d_in[1];
    const float* w_out = (const float*)d_in[2];
    const float* b_out = (const float*)d_in[3];
    const int*   mask  = (const int*)d_in[4];
    float*       out   = (float*)d_out;

    const size_t x_elems   = (size_t)BATCH * SEQ * DMODEL;      // 8388608
    const size_t qkv_elems = (size_t)BATCH * NHEAD * SEQ * DHEAD;

    bf16_t* wqkvT = (bf16_t*)d_ws;
    bf16_t* woutT = wqkvT + (size_t)QKVC * DMODEL;
    bf16_t* xb    = woutT + (size_t)DMODEL * DMODEL;
    bf16_t* Qb    = xb + x_elems;
    bf16_t* Kb    = Qb + qkv_elems;
    bf16_t* Vt    = Kb + qkv_elems;
    bf16_t* aout  = Vt + qkv_elems;

    convert_f32_bf16<<<(int)(x_elems / 8 / 256), 256, 0, stream>>>(
        x, xb, (int)(x_elems / 8));
    transpose_to_bf16<<<(DMODEL / 32) * (QKVC / 32), 256, 0, stream>>>(
        w_qkv, wqkvT, DMODEL, QKVC);
    transpose_to_bf16<<<(DMODEL / 32) * (DMODEL / 32), 256, 0, stream>>>(
        w_out, woutT, DMODEL, DMODEL);

    qkv_gemm<<<64 * (QKVC / 128), 256, 0, stream>>>(xb, wqkvT, Qb, Kb, Vt);

    flash_attn<<<BATCH * NHEAD * (SEQ / 64), 128, 0, stream>>>(
        Qb, Kb, Vt, mask, aout);

    out_gemm<<<64 * (DMODEL / 128), 256, 0, stream>>>(aout, woutT, b_out, out);
}
